// LIDAR2D_28870770164157
// MI455X (gfx1250) — matrix operand order repaired and checked by reading, not compiled, not run on hardware
//
#include <hip/hip_runtime.h>
#include <cstdint>
#include <cstddef>

// ---------------------------------------------------------------------------
// CDNA5 (gfx1250, wave32) 4-direction 2D selective-scan block.
// Dense GEMMs: v_wmma_f32_16x16x32_bf16. A operands stream from global as
// contiguous b128 loads; B panels are staged per-block into LDS and read as
// ds_load_b128, shared by the 8 waves of the block. k_out stages its panel
// with the Tensor Data Mover (tensor_load_to_lds + s_wait_tensorcnt), using
// the D# pad feature to reproduce the bank-rotation row padding in hardware.
// The sequential scan is parallelized over (batch, channel) lanes with
// per-lane register state and global_prefetch lookahead.
// ---------------------------------------------------------------------------

typedef __bf16 bf16_t;
typedef __attribute__((ext_vector_type(16))) __bf16 v16bf;
typedef __attribute__((ext_vector_type(8)))  __bf16 v8bf;
typedef __attribute__((ext_vector_type(8)))  float  v8f;
typedef __attribute__((ext_vector_type(4)))  float  v4f;
typedef __attribute__((ext_vector_type(4)))  unsigned int u32x4;
typedef __attribute__((ext_vector_type(8)))  int i32x8;
typedef __attribute__((ext_vector_type(4)))  int i32x4;

#define DEV static __device__ __forceinline__

// Problem constants (fixed by setup_inputs)
constexpr int BATCH = 2;
constexpr int HH    = 48;
constexpr int WW    = 48;
constexpr int LSEQ  = HH * WW;          // 2304
constexpr int DM    = 256;
constexpr int EE    = 512;
constexpr int TWOE  = 1024;
constexpr int MID   = 32;
constexpr int NN    = 16;
constexpr int RR    = 16;
constexpr int XP    = 48;               // R + 2N
constexpr int MROWS = BATCH * LSEQ;     // 4608 rows in all GEMMs

constexpr int KC  = 256;                // K staging chunk
constexpr int PAD = 8;                  // LDS row pad (bf16 elems) = 16B
constexpr int LDR = KC + PAD;           // LDS row stride (264 bf16)

DEV bf16_t to_bf16(float f) {
  union { float f; uint32_t u; } v; v.f = f;
  uint32_t u = v.u;
  uint32_t r = (u + 0x7FFFu + ((u >> 16) & 1u)) >> 16;   // round-to-nearest-even
  unsigned short s = (unsigned short)r;
  return __builtin_bit_cast(bf16_t, s);
}

DEV float siluf(float v) { return v / (1.0f + expf(-v)); }
DEV float softplusf(float v) { return (v > 20.0f) ? v : log1pf(expf(v)); }

DEV v16bf cat16(v8bf lo, v8bf hi) {
  return __builtin_shufflevector(lo, hi, 0, 1, 2, 3, 4, 5, 6, 7,
                                 8, 9, 10, 11, 12, 13, 14, 15);
}

// A fragment 16x32 bf16 (MxK), source row-major [m][ldk]. Lane m = lane&15;
// hi lane-group holds K 8..15 / 24..31 (ISA "16-bit A-Matrix 16x32").
DEV v16bf load_afrag(const bf16_t* __restrict__ base, int ldk, int lane) {
  const int m  = lane & 15;
  const int hb = (lane >> 4) << 3;
  const bf16_t* p = base + (size_t)m * ldk + hb;
  v8bf lo = *(const v8bf*)p;          // K hb..hb+7      (16B load)
  v8bf hi = *(const v8bf*)(p + 16);   // K 16+hb..16+hb+7
  return cat16(lo, hi);
}

// B fragment 32x16 bf16 (KxN), source K-major [n][ldk]. Lane n = lane&15;
// hi lane-group holds K 8..15 / 24..31 (same as A). Works for global or LDS
// source pointers (addrspace inferred -> global_load_b128 / ds_load_b128).
DEV v16bf load_bfrag(const bf16_t* base, int ldk, int lane) {
  const int n  = lane & 15;
  const int kb = (lane >> 4) << 3;
  const bf16_t* p = base + (size_t)n * ldk + kb;
  v8bf lo = *(const v8bf*)p;          // 16B load
  v8bf hi = *(const v8bf*)(p + 16);   // 16B load
  return cat16(lo, hi);
}

DEV v8f wmma_bf16(v16bf a, v16bf b, v8f c) {
  return __builtin_amdgcn_wmma_f32_16x16x32_bf16(false, a, false, b, (short)0, c,
                                                 false, false);
}

DEV void zero_acc(v8f* acc, int n) {
  for (int t = 0; t < n; ++t)
#pragma unroll
    for (int r = 0; r < 8; ++r) acc[t][r] = 0.0f;
}

// Cooperative LDS fill: COLS rows x KC bf16 from B[n0+n][kc..kc+KC) (K-major).
DEV void fill_panel(bf16_t* Bs, const bf16_t* __restrict__ Bg, int ldk,
                    int n0, int kc, int cols, int tid) {
  const int chunks = cols * (KC / 8);
  for (int c = tid; c < chunks; c += 256) {
    const int n  = c >> 5;               // KC/8 == 32 chunks per row
    const int k8 = (c & 31) << 3;
    *(v8bf*)(Bs + n * LDR + k8) =
        *(const v8bf*)(Bg + (size_t)(n0 + n) * ldk + kc + k8);
  }
}

// TDM panel fill: issue one tensor_load_to_lds (wave-scalar) for a 2D tile of
// `cols` rows x KC bf16, with hardware LDS row padding of 16B every 512B
// (pad_interval=6 -> 128 DWORDs, pad_amount=3 -> 4 DWORDs) matching LDR.
#if __has_builtin(__builtin_amdgcn_tensor_load_to_lds)
#define HAVE_TDM 1
DEV void tdm_fill_panel(bf16_t* Bs, const bf16_t* Bg, int ldk,
                        int n0, int kc, int cols) {
  const uint64_t gaddr = (uint64_t)(uintptr_t)(Bg + (size_t)n0 * ldk + kc);
  // LDS aperture: low 32 bits of the flat address are the LDS byte address.
  const uint32_t laddr = (uint32_t)(uintptr_t)(void*)Bs;

  u32x4 g0;
  g0[0] = 1u;                                         // count=1, user D#
  g0[1] = laddr;                                      // lds_addr
  g0[2] = (uint32_t)gaddr;                            // global_addr[31:0]
  g0[3] = (uint32_t)((gaddr >> 32) & 0x1FFFFFFu)      // global_addr[56:32]
        | (2u << 30);                                 // type=2 ("image")
  i32x8 g1;
  g1[0] = (1 << 16)                                   // data_size = 2 bytes
        | (1 << 20)                                   // pad_enable
        | (6 << 22)                                   // pad_interval: 128 DW
        | (3 << 25);                                  // pad_amount: 4 DW (16B)
  g1[1] = (ldk & 0xFFFF) << 16;                       // tensor_dim0[15:0]
  g1[2] = ((ldk >> 16) & 0xFFFF) | (cols << 16);      // dim0[31:16], dim1[15:0]
  g1[3] = (KC << 16);                                 // tile_dim0 = 256
  g1[4] = cols;                                       // tile_dim1; tile_dim2=0
  g1[5] = ldk;                                        // tensor_dim0_stride
  g1[6] = 0;                                          // stride hi, dim1_stride
  g1[7] = 0;
  i32x4 g2 = {0, 0, 0, 0};                            // unused (2D tile)
  i32x4 g3 = {0, 0, 0, 0};
#if defined(__clang_major__) && __clang_major__ >= 23
  i32x8 g4 = {0, 0, 0, 0, 0, 0, 0, 0};
  __builtin_amdgcn_tensor_load_to_lds(g0, g1, g2, g3, g4, 0);
#else
  __builtin_amdgcn_tensor_load_to_lds(g0, g1, g2, g3, 0);
#endif
  __builtin_amdgcn_s_wait_tensorcnt(0);               // issuing wave drains TDM
}
#else
#define HAVE_TDM 0
#endif

// ---------------------------------------------------------------------------
// Prep kernels: one-shot bf16 conversion / transposition of GEMM operands.
// ---------------------------------------------------------------------------

// xin[m][k] = bf16(x[m][k] + pos[l][k]); 8 elements per thread, b128 store.
__global__ void k_prep_xin(const float* __restrict__ x, const float* __restrict__ W_pos,
                           const float* __restrict__ b_pos, bf16_t* __restrict__ xin) {
  const int idx = blockIdx.x * blockDim.x + threadIdx.x;   // < MROWS*256/8
  const int m  = idx >> 5;
  const int k0 = (idx & 31) << 3;
  const int l  = m % LSEQ;
  const float ci = (float)(l / WW) * (2.0f / 47.0f) - 1.0f;
  const float cj = (float)(l % WW) * (2.0f / 47.0f) - 1.0f;
  v8bf o;
#pragma unroll
  for (int j = 0; j < 8; ++j) {
    const int kk = k0 + j;
    o[j] = to_bf16(x[(size_t)m * DM + kk] + ci * W_pos[kk] + cj * W_pos[DM + kk] + b_pos[kk]);
  }
  *(v8bf*)(xin + (size_t)m * DM + k0) = o;
}

__global__ void k_cvt(const float* __restrict__ src, bf16_t* __restrict__ dst, int count) {
  const int idx = blockIdx.x * blockDim.x + threadIdx.x;
  if (idx < count) dst[idx] = to_bf16(src[idx]);
}

// dst[n*K + k] = bf16(src[k*N + n])   (transpose to K-major for B fragments)
__global__ void k_cvtT(const float* __restrict__ src, bf16_t* __restrict__ dst,
                       int K, int N) {
  const int idx = blockIdx.x * blockDim.x + threadIdx.x;
  if (idx >= K * N) return;
  const int n = idx / K, k = idx % K;
  dst[idx] = to_bf16(src[(size_t)k * N + n]);
}

// dst[n*Kpad + k] = k<K ? bf16(src[k*N + n]) : 0    (W_dt: K 16 -> 32 pad)
__global__ void k_cvtT_pad(const float* __restrict__ src, bf16_t* __restrict__ dst,
                           int K, int N, int Kpad) {
  const int idx = blockIdx.x * blockDim.x + threadIdx.x;
  if (idx >= Kpad * N) return;
  const int n = idx / Kpad, k = idx % Kpad;
  dst[idx] = (k < K) ? to_bf16(src[(size_t)k * N + n]) : to_bf16(0.0f);
}

// ---------------------------------------------------------------------------
// K1: xz = xin @ W_in         (M=4608,K=256,N=1024) -> xh (bf16), z (f32)
// Block: 8 waves = 8 M-tiles sharing one 64-col B panel staged in LDS.
// ---------------------------------------------------------------------------
__global__ void k_gemm_xz(const bf16_t* __restrict__ xin, const bf16_t* __restrict__ WtIn,
                          bf16_t* __restrict__ xh, float* __restrict__ z) {
  __shared__ bf16_t Bs[64 * LDR];
  const int tid  = threadIdx.x;
  const int lane = tid & 31;
  const int wv   = tid >> 5;
  const int ngrp = blockIdx.x & 15;                 // 16 col-groups of 64
  const int m0   = ((blockIdx.x >> 4) * 8 + wv) * 16;
  const int n0   = ngrp * 64;

  v8f acc[4]; zero_acc(acc, 4);
  fill_panel(Bs, WtIn, DM, n0, 0, 64, tid);
  __syncthreads();
#pragma unroll
  for (int k0 = 0; k0 < DM; k0 += 32) {
    v16bf a = load_afrag(xin + (size_t)m0 * DM + k0, DM, lane);
#pragma unroll
    for (int t = 0; t < 4; ++t) {
      v16bf b = load_bfrag(Bs + (t * 16) * LDR + k0, LDR, lane);
      acc[t] = wmma_bf16(a, b, acc[t]);
    }
  }
  const int n = lane & 15, mloc = (lane >> 4) << 3;
  if (n0 < EE) {   // xh half -> bf16 (GEMM A operand of pw1)
#pragma unroll
    for (int t = 0; t < 4; ++t)
#pragma unroll
      for (int r = 0; r < 8; ++r)
        xh[(size_t)(m0 + mloc + r) * EE + n0 + t * 16 + n] = to_bf16(acc[t][r]);
  } else {         // z half -> f32 (combine kernel)
#pragma unroll
    for (int t = 0; t < 4; ++t)
#pragma unroll
      for (int r = 0; r < 8; ++r)
        z[(size_t)(m0 + mloc + r) * EE + (n0 - EE) + t * 16 + n] = acc[t][r];
  }
}

// ---------------------------------------------------------------------------
// K2: h1 = xh @ pw1_w^T + pw1_b      (M=4608,K=512,N=32) -> f32
// ---------------------------------------------------------------------------
__global__ void k_pw1(const bf16_t* __restrict__ xh, const bf16_t* __restrict__ pw1w,
                      const float* __restrict__ pw1_b, float* __restrict__ h1) {
  __shared__ bf16_t Bs[32 * LDR];
  const int tid  = threadIdx.x;
  const int lane = tid & 31;
  const int m0   = (blockIdx.x * 8 + (tid >> 5)) * 16;

  v8f acc[2]; zero_acc(acc, 2);
  for (int kc = 0; kc < EE; kc += KC) {
    fill_panel(Bs, pw1w, EE, 0, kc, 32, tid);
    __syncthreads();
#pragma unroll
    for (int k0 = 0; k0 < KC; k0 += 32) {
      v16bf a = load_afrag(xh + (size_t)m0 * EE + kc + k0, EE, lane);
#pragma unroll
      for (int t = 0; t < 2; ++t) {
        v16bf b = load_bfrag(Bs + (t * 16) * LDR + k0, LDR, lane);
        acc[t] = wmma_bf16(a, b, acc[t]);
      }
    }
    __syncthreads();
  }
  const int n = lane & 15, mloc = (lane >> 4) << 3;
#pragma unroll
  for (int t = 0; t < 2; ++t) {
    const float bias = pw1_b[t * 16 + n];
#pragma unroll
    for (int r = 0; r < 8; ++r)
      h1[(size_t)(m0 + mloc + r) * MID + t * 16 + n] = acc[t][r] + bias;
  }
}

// ---------------------------------------------------------------------------
// K3: depthwise 3x3, zero pad, NHWC(32) -> h2 (bf16)
// ---------------------------------------------------------------------------
__global__ void k_dw(const float* __restrict__ h1, const float* __restrict__ dw_w,
                     bf16_t* __restrict__ h2) {
  const int idx = blockIdx.x * blockDim.x + threadIdx.x;  // < 4608*32
  const int c = idx & 31;
  const int m = idx >> 5;
  const int b = m / LSEQ, l = m % LSEQ;
  const int i = l / WW, j = l % WW;
  float s = 0.0f;
#pragma unroll
  for (int di = 0; di < 3; ++di) {
    const int ii = i + di - 1;
    if ((unsigned)ii >= (unsigned)HH) continue;
#pragma unroll
    for (int dj = 0; dj < 3; ++dj) {
      const int jj = j + dj - 1;
      if ((unsigned)jj >= (unsigned)WW) continue;
      s += h1[(size_t)(b * LSEQ + ii * WW + jj) * MID + c] * dw_w[c * 9 + di * 3 + dj];
    }
  }
  h2[(size_t)m * MID + c] = to_bf16(s);
}

// ---------------------------------------------------------------------------
// K4: x_conv = silu(h2 @ pw2_w^T)   (M=4608,K=32,N=512) -> f32 + bf16
// B tiny (single k-step): direct global fragments.
// ---------------------------------------------------------------------------
__global__ void k_pw2(const bf16_t* __restrict__ h2, const bf16_t* __restrict__ pw2w,
                      float* __restrict__ xconv, bf16_t* __restrict__ xconv_bf) {
  const int lane = threadIdx.x & 31;
  const int gw = blockIdx.x * (blockDim.x >> 5) + (threadIdx.x >> 5);  // 0..2303
  const int m0 = (gw >> 3) * 16;
  const int n0 = (gw & 7) * 64;

  v8f acc[4]; zero_acc(acc, 4);
  v16bf a = load_afrag(h2 + (size_t)m0 * MID, MID, lane);
#pragma unroll
  for (int t = 0; t < 4; ++t) {
    v16bf b = load_bfrag(pw2w + (size_t)(n0 + t * 16) * MID, MID, lane);
    acc[t] = wmma_bf16(a, b, acc[t]);
  }
  const int n = lane & 15, mloc = (lane >> 4) << 3;
#pragma unroll
  for (int t = 0; t < 4; ++t)
#pragma unroll
    for (int r = 0; r < 8; ++r) {
      const float v = siluf(acc[t][r]);
      const size_t o = (size_t)(m0 + mloc + r) * EE + n0 + t * 16 + n;
      xconv[o]    = v;
      xconv_bf[o] = to_bf16(v);
    }
}

// ---------------------------------------------------------------------------
// K5: x_dbl = x_conv @ W_xproj      (M=4608,K=512,N=48) -> f32; dt_r -> bf16
// ---------------------------------------------------------------------------
__global__ void k_xdbl(const bf16_t* __restrict__ xconv_bf, const bf16_t* __restrict__ Wxp,
                       float* __restrict__ xdbl, bf16_t* __restrict__ dtr_bf) {
  __shared__ bf16_t Bs[48 * LDR];
  const int tid  = threadIdx.x;
  const int lane = tid & 31;
  const int m0   = (blockIdx.x * 8 + (tid >> 5)) * 16;

  v8f acc[3]; zero_acc(acc, 3);
  for (int kc = 0; kc < EE; kc += KC) {
    fill_panel(Bs, Wxp, EE, 0, kc, 48, tid);
    __syncthreads();
#pragma unroll
    for (int k0 = 0; k0 < KC; k0 += 32) {
      v16bf a = load_afrag(xconv_bf + (size_t)m0 * EE + kc + k0, EE, lane);
#pragma unroll
      for (int t = 0; t < 3; ++t) {
        v16bf b = load_bfrag(Bs + (t * 16) * LDR + k0, LDR, lane);
        acc[t] = wmma_bf16(a, b, acc[t]);
      }
    }
    __syncthreads();
  }
  const int n = lane & 15, mloc = (lane >> 4) << 3;
#pragma unroll
  for (int t = 0; t < 3; ++t)
#pragma unroll
    for (int r = 0; r < 8; ++r)
      xdbl[(size_t)(m0 + mloc + r) * XP + t * 16 + n] = acc[t][r];
#pragma unroll
  for (int r = 0; r < 8; ++r) {               // dt_r bf16 copy, K-padded to 32
    const size_t row = (size_t)(m0 + mloc + r) * MID;
    dtr_bf[row + n]      = to_bf16(acc[0][r]);
    dtr_bf[row + 16 + n] = to_bf16(0.0f);
  }
}

// ---------------------------------------------------------------------------
// K6: delta = softplus(dt_r @ W_dt + 2*b_dt)   (M=4608,K=32(pad),N=512)
// ---------------------------------------------------------------------------
__global__ void k_dt(const bf16_t* __restrict__ dtr_bf, const bf16_t* __restrict__ Wdt,
                     const float* __restrict__ b_dt, float* __restrict__ delta) {
  const int lane = threadIdx.x & 31;
  const int gw = blockIdx.x * (blockDim.x >> 5) + (threadIdx.x >> 5);  // 0..2303
  const int m0 = (gw >> 3) * 16;
  const int n0 = (gw & 7) * 64;

  v8f acc[4]; zero_acc(acc, 4);
  v16bf a = load_afrag(dtr_bf + (size_t)m0 * MID, MID, lane);
#pragma unroll
  for (int t = 0; t < 4; ++t) {
    v16bf b = load_bfrag(Wdt + (size_t)(n0 + t * 16) * MID, MID, lane);
    acc[t] = wmma_bf16(a, b, acc[t]);
  }
  const int n = lane & 15, mloc = (lane >> 4) << 3;
#pragma unroll
  for (int t = 0; t < 4; ++t) {
    const float b2 = 2.0f * b_dt[n0 + t * 16 + n];   // reference adds b_dt twice
#pragma unroll
    for (int r = 0; r < 8; ++r)
      delta[(size_t)(m0 + mloc + r) * EE + n0 + t * 16 + n] = softplusf(acc[t][r] + b2);
  }
}

// ---------------------------------------------------------------------------
// K7: 4-direction selective scan. One lane per (b,e); h[4][16] in VGPRs.
// Coeffs (dA/dB/C) shared across directions; u permuted via closed forms.
// ---------------------------------------------------------------------------
__global__ void k_scan(const float* __restrict__ delta, const float* __restrict__ xdbl,
                       const float* __restrict__ xconv, const float* __restrict__ A_log,
                       const float* __restrict__ dir_emb, float* __restrict__ ys) {
  const int g = blockIdx.x * blockDim.x + threadIdx.x;   // 0..1023
  const int b = g >> 9;                                  // uniform per block
  const int e = g & 511;

  float A[NN];
#pragma unroll
  for (int n = 0; n < NN; ++n) A[n] = -expf(A_log[e * NN + n]);
  const float de0 = dir_emb[e];
  const float de1 = dir_emb[EE + e];
  const float de2 = dir_emb[2 * EE + e];
  const float de3 = dir_emb[3 * EE + e];

  float h[4][NN];
#pragma unroll
  for (int k = 0; k < 4; ++k)
#pragma unroll
    for (int n = 0; n < NN; ++n) h[k][n] = 0.0f;

  const float* xc = xconv + (size_t)b * LSEQ * EE + e;
  const float* dl = delta + (size_t)b * LSEQ * EE + e;

  for (int t = 0; t < LSEQ; ++t) {
    if (t + 16 < LSEQ) {                                 // gfx1250 global_prefetch
      __builtin_prefetch(dl + (size_t)(t + 16) * EE, 0, 1);
      __builtin_prefetch(xc + (size_t)(t + 16) * EE, 0, 1);
    }
    const float dv = dl[(size_t)t * EE];
    const float* bc = xdbl + (size_t)(b * LSEQ + t) * XP;  // block-uniform address
    v4f Bv[4], Cv[4];
#pragma unroll
    for (int q = 0; q < 4; ++q) {                        // 16B vector loads
      Bv[q] = *(const v4f*)(bc + RR + 4 * q);
      Cv[q] = *(const v4f*)(bc + RR + NN + 4 * q);
    }

    const int l1 = (LSEQ - 1) - t;
    const int l2 = (t % WW) * WW + t / WW;
    const int l3 = (l1 % WW) * WW + l1 / WW;
    const float u0 = xc[(size_t)t  * EE] + de0;
    const float u1 = xc[(size_t)l1 * EE] + de1;
    const float u2 = xc[(size_t)l2 * EE] + de2;
    const float u3 = xc[(size_t)l3 * EE] + de3;

    float y0 = 0.0f, y1 = 0.0f, y2 = 0.0f, y3 = 0.0f;
#pragma unroll
    for (int n = 0; n < NN; ++n) {
      const float dA = __expf(dv * A[n]);
      const float dB = dv * Bv[n >> 2][n & 3];
      const float C  = Cv[n >> 2][n & 3];
      h[0][n] = dA * h[0][n] + dB * u0; y0 += h[0][n] * C;
      h[1][n] = dA * h[1][n] + dB * u1; y1 += h[1][n] * C;
      h[2][n] = dA * h[2][n] + dB * u2; y2 += h[2][n] * C;
      h[3][n] = dA * h[3][n] + dB * u3; y3 += h[3][n] * C;
    }
    const size_t base = ((size_t)b * LSEQ + t) * EE + e;
    const size_t kstr = (size_t)2 * LSEQ * EE;
    ys[base]            = y0;
    ys[base + kstr]     = y1;
    ys[base + 2 * kstr] = y2;
    ys[base + 3 * kstr] = y3;
  }
}

// ---------------------------------------------------------------------------
// K8: gather inverse permutations, add u*Dp skip, multiply silu(z) -> bf16
// ---------------------------------------------------------------------------
__global__ void k_combine(const float* __restrict__ ys, const float* __restrict__ xconv,
                          const float* __restrict__ dir_emb, const float* __restrict__ Dp,
                          const float* __restrict__ z, bf16_t* __restrict__ ycomb) {
  const size_t idx = (size_t)blockIdx.x * blockDim.x + threadIdx.x;  // < 4608*512
  const int e = (int)(idx & 511);
  const int m = (int)(idx >> 9);
  const int b = m / LSEQ, l = m % LSEQ;

  const int t0 = l;
  const int t1 = (LSEQ - 1) - l;
  const int t2 = (l % WW) * WW + l / WW;       // o3 is an involution
  const int t3 = (LSEQ - 1) - t2;

  const size_t kstr = (size_t)2 * LSEQ * EE;
  const size_t rowb = (size_t)b * LSEQ * EE + (size_t)e;
  float s = ys[rowb + (size_t)t0 * EE]
          + ys[rowb + (size_t)t1 * EE + kstr]
          + ys[rowb + (size_t)t2 * EE + 2 * kstr]
          + ys[rowb + (size_t)t3 * EE + 3 * kstr];
  const float de = dir_emb[e] + dir_emb[EE + e] + dir_emb[2 * EE + e] + dir_emb[3 * EE + e];
  s += (4.0f * xconv[(size_t)m * EE + e] + de) * Dp[e];
  const float zz = z[(size_t)m * EE + e];
  ycomb[(size_t)m * EE + e] = to_bf16(s * siluf(zz));
}

// ---------------------------------------------------------------------------
// K9: out = ycomb @ W_out            (M=4608,K=512,N=256) -> f32
// B panel staged via Tensor Data Mover (falls back to cooperative fill).
// ---------------------------------------------------------------------------
__global__ void k_out(const bf16_t* __restrict__ ycomb, const bf16_t* __restrict__ Wout,
                      float* __restrict__ out) {
  __shared__ bf16_t Bs[64 * LDR];
  const int tid  = threadIdx.x;
  const int lane = tid & 31;
  const int ngrp = blockIdx.x & 3;                // 4 col-groups of 64
  const int m0   = ((blockIdx.x >> 2) * 8 + (tid >> 5)) * 16;
  const int n0   = ngrp * 64;

  v8f acc[4]; zero_acc(acc, 4);
  for (int kc = 0; kc < EE; kc += KC) {
#if HAVE_TDM
    if (tid < 32) {                               // wave 0 drives the TDM
      tdm_fill_panel(Bs, Wout, EE, n0, kc, 64);
    }
#else
    fill_panel(Bs, Wout, EE, n0, kc, 64, tid);
#endif
    __syncthreads();
#pragma unroll
    for (int k0 = 0; k0 < KC; k0 += 32) {
      v16bf a = load_afrag(ycomb + (size_t)m0 * EE + kc + k0, EE, lane);
#pragma unroll
      for (int t = 0; t < 4; ++t) {
        v16bf b = load_bfrag(Bs + (t * 16) * LDR + k0, LDR, lane);
        acc[t] = wmma_bf16(a, b, acc[t]);
      }
    }
    __syncthreads();
  }
  const int n = lane & 15, mloc = (lane >> 4) << 3;
#pragma unroll
  for (int t = 0; t < 4; ++t)
#pragma unroll
    for (int r = 0; r < 8; ++r)
      out[(size_t)(m0 + mloc + r) * DM + n0 + t * 16 + n] = acc[t][r];
}

// ---------------------------------------------------------------------------
extern "C" void kernel_launch(void* const* d_in, const int* in_sizes, int n_in,
                              void* d_out, int out_size, void* d_ws, size_t ws_size,
                              hipStream_t stream) {
  (void)in_sizes; (void)n_in; (void)out_size; (void)ws_size;

  const float* x       = (const float*)d_in[0];
  const float* W_pos   = (const float*)d_in[1];
  const float* b_pos   = (const float*)d_in[2];
  const float* W_in    = (const float*)d_in[3];
  const float* pw1_w   = (const float*)d_in[4];
  const float* pw1_b   = (const float*)d_in[5];
  const float* dw_w    = (const float*)d_in[6];
  const float* pw2_w   = (const float*)d_in[7];
  const float* W_xproj = (const float*)d_in[8];
  const float* W_dt    = (const float*)d_in[9];
  const float* b_dt    = (const float*)d_in[10];
  const float* A_log   = (const float*)d_in[11];
  const float* Dp      = (const float*)d_in[12];
  const float* dir_emb = (const float*)d_in[13];
  const float* W_out   = (const float*)d_in[14];
  float* out = (float*)d_out;

  const size_t M = (size_t)MROWS;
  char* wp = (char*)d_ws;
  auto alloc = [&](size_t bytes) -> char* {
    char* r = wp; wp += (bytes + 255) & ~(size_t)255; return r;
  };

  bf16_t* xin      = (bf16_t*)alloc(M * DM * 2);          // A of K1
  bf16_t* WtIn     = (bf16_t*)alloc((size_t)TWOE * DM * 2);
  bf16_t* pw1w_bf  = (bf16_t*)alloc((size_t)MID * EE * 2);
  bf16_t* pw2w_bf  = (bf16_t*)alloc((size_t)EE * MID * 2);
  bf16_t* Wxp_bf   = (bf16_t*)alloc((size_t)XP * EE * 2);
  bf16_t* Wdt_bf   = (bf16_t*)alloc((size_t)EE * MID * 2); // K padded 16->32
  bf16_t* Wout_bf  = (bf16_t*)alloc((size_t)DM * EE * 2);
  bf16_t* xh       = (bf16_t*)alloc(M * EE * 2);
  float*  z        = (float*)alloc(M * EE * 4);
  float*  h1       = (float*)alloc(M * MID * 4);
  bf16_t* h2       = (bf16_t*)alloc(M * MID * 2);
  float*  xconv    = (float*)alloc(M * EE * 4);
  bf16_t* xconv_bf = (bf16_t*)alloc(M * EE * 2);
  float*  xdbl     = (float*)alloc(M * XP * 4);
  bf16_t* dtr_bf   = (bf16_t*)alloc(M * MID * 2);
  float*  delta    = (float*)alloc(M * EE * 4);
  float*  ys       = (float*)alloc(4 * M * EE * 4);
  bf16_t* ycomb    = (bf16_t*)alloc(M * EE * 2);

  // --- operand prep (independent) ---
  k_prep_xin<<<576, 256, 0, stream>>>(x, W_pos, b_pos, xin);
  k_cvtT    <<<1024, 256, 0, stream>>>(W_in, WtIn, DM, TWOE);
  k_cvt     <<<64, 256, 0, stream>>>(pw1_w, pw1w_bf, MID * EE);
  k_cvt     <<<64, 256, 0, stream>>>(pw2_w, pw2w_bf, EE * MID);
  k_cvtT    <<<96, 256, 0, stream>>>(W_xproj, Wxp_bf, EE, XP);
  k_cvtT_pad<<<64, 256, 0, stream>>>(W_dt, Wdt_bf, RR, EE, MID);
  k_cvtT    <<<512, 256, 0, stream>>>(W_out, Wout_bf, EE, DM);

  // --- pipeline ---
  k_gemm_xz<<<576, 256, 0, stream>>>(xin, WtIn, xh, z);
  k_pw1    <<<36,  256, 0, stream>>>(xh, pw1w_bf, pw1_b, h1);
  k_dw     <<<576, 256, 0, stream>>>(h1, dw_w, h2);
  k_pw2    <<<288, 256, 0, stream>>>(h2, pw2w_bf, xconv, xconv_bf);
  k_xdbl   <<<36,  256, 0, stream>>>(xconv_bf, Wxp_bf, xdbl, dtr_bf);
  k_dt     <<<288, 256, 0, stream>>>(dtr_bf, Wdt_bf, b_dt, delta);
  k_scan   <<<4,   256, 0, stream>>>(delta, xdbl, xconv, A_log, dir_emb, ys);
  k_combine<<<9216,256, 0, stream>>>(ys, xconv, dir_emb, Dp, z, ycomb);
  k_out    <<<144, 256, 0, stream>>>(ycomb, Wout_bf, out);
}
